// FutoshikiGGCNN_16123307229949
// MI455X (gfx1250) — matrix operand order repaired and checked by reading, not compiled, not run on hardware
//
#include <hip/hip_runtime.h>

typedef _Float16 half_t;
typedef __attribute__((ext_vector_type(16))) _Float16 v16h;
typedef __attribute__((ext_vector_type(8)))  _Float16 v8h;
typedef __attribute__((ext_vector_type(8)))  float    v8f;

#define HD    128
#define NCELL 16384
#define NCLU  1088
#define NB    64
#define ET    131072
#define NT    5
#define NS    3

// ---------------- WMMA fragment helpers (wave32; contiguous k = 16*hi + s, same order in A and B) ----------------

// A fragment (16x32 f16, M x K): lane lo = row, 16 contiguous k starting at k0 + hi*16
__device__ inline v16h frag_a16(const half_t* base, int ld, int m0, int k0, int lo, int hi) {
    const half_t* r = base + (size_t)(m0 + lo) * ld + k0 + hi * 16;
    v16h o = *(const v16h*)r;
    return o;
}

// B fragment (32x16 f16, K x N) from N-major (transposed) weights:
// lane lo = column n, 16 contiguous k starting at k0 + hi*16
__device__ inline v16h frag_b16(const half_t* base, int ld, int n0, int k0, int lo, int hi) {
    return *(const v16h*)(base + (size_t)(n0 + lo) * ld + k0 + hi * 16);
}

__device__ inline v8f wmma_f16(v16h a, v16h b, v8f c) {
    return __builtin_amdgcn_wmma_f32_16x16x32_f16(false, a, false, b, (short)0, c, false, false);
}

__device__ inline float sigm(float x) { return 1.0f / (1.0f + __expf(-x)); }

// ---------------- Edge-type fused 4-layer MLP + scatter-add ----------------
// One block per 128 edges. A0 = [hs | hd] (128 x 256 f16) in LDS; weights staged
// in LDS; z intermediates double-buffered in the A0 region. Scatter via fp32 atomics.

__global__ __launch_bounds__(256) void edge_mlp_kernel(
    const int* __restrict__ src, const int* __restrict__ dst,
    const half_t* __restrict__ hsrc, const half_t* __restrict__ hcell,
    const half_t* __restrict__ W1t, const half_t* __restrict__ W2t,
    const half_t* __restrict__ W3t, const half_t* __restrict__ W4t,
    const float* __restrict__ b1, const float* __restrict__ b2,
    const float* __restrict__ b3, const float* __restrict__ b4,
    float* __restrict__ msg)
{
    __shared__ __align__(64) half_t lds_w[256 * HD];   // 64 KB weight stage
    __shared__ __align__(64) half_t lds_a[128 * 256];  // 64 KB A0 / z buffers
    half_t* zbuf0 = lds_a;
    half_t* zbuf1 = lds_a + 128 * HD;

    const int tid  = threadIdx.x;
    const int wave = tid >> 5;
    const int lane = tid & 31;
    const int lo   = lane & 15;
    const int hi   = lane >> 4;
    const int e0   = blockIdx.x * 128;

    // gather A0: thread -> (edge row, hs/hd half); 256 B contiguous f16 copy each
    {
        const int r   = tid >> 1;
        const int sel = tid & 1;
        const int idx = sel ? dst[e0 + r] : src[e0 + r];
        const int4* s4 = (const int4*)((sel ? hcell : hsrc) + (size_t)idx * HD);
        int4* d4 = (int4*)(lds_a + r * 256 + sel * HD);
#pragma unroll
        for (int i = 0; i < 16; ++i) d4[i] = s4[i];
    }
    // stage W1t (128 x 256 f16, N-major)
    {
        const int4* s4 = (const int4*)W1t;
        int4* d4 = (int4*)lds_w;
#pragma unroll
        for (int i = 0; i < 16; ++i) d4[tid + 256 * i] = s4[tid + 256 * i];
    }
    __syncthreads();

    v8f acc[8];

    // ---- layer 1: [128x256] @ [256x128] ----
#pragma unroll
    for (int j = 0; j < 8; ++j)
#pragma unroll
        for (int r = 0; r < 8; ++r) acc[j][r] = 0.f;
    for (int kk = 0; kk < 256; kk += 32) {
        v16h a = frag_a16(lds_a, 256, wave * 16, kk, lo, hi);
#pragma unroll
        for (int j = 0; j < 8; ++j) {
            v16h b = frag_b16(lds_w, 256, j * 16, kk, lo, hi);
            acc[j] = wmma_f16(a, b, acc[j]);
        }
    }
    __syncthreads();
    // relu epilogue -> zbuf0 ; stage W2
#pragma unroll
    for (int j = 0; j < 8; ++j)
#pragma unroll
        for (int r = 0; r < 8; ++r) {
            int m = wave * 16 + r + hi * 8;
            int n = j * 16 + lo;
            float v = acc[j][r] + b1[n];
            zbuf0[m * HD + n] = (half_t)fmaxf(v, 0.f);
        }
    {
        const int4* s4 = (const int4*)W2t;
        int4* d4 = (int4*)lds_w;
#pragma unroll
        for (int i = 0; i < 8; ++i) d4[tid + 256 * i] = s4[tid + 256 * i];
    }
    __syncthreads();

    // ---- layers 2 and 3 (relu) ----
    const half_t* stageW[2] = { W3t, W4t };
    const float*  biasL[2]  = { b2, b3 };
    half_t* zin  = zbuf0;
    half_t* zout = zbuf1;
    for (int L = 0; L < 2; ++L) {
#pragma unroll
        for (int j = 0; j < 8; ++j)
#pragma unroll
            for (int r = 0; r < 8; ++r) acc[j][r] = 0.f;
        for (int kk = 0; kk < HD; kk += 32) {
            v16h a = frag_a16(zin, HD, wave * 16, kk, lo, hi);
#pragma unroll
            for (int j = 0; j < 8; ++j) {
                v16h b = frag_b16(lds_w, HD, j * 16, kk, lo, hi);
                acc[j] = wmma_f16(a, b, acc[j]);
            }
        }
        __syncthreads();
#pragma unroll
        for (int j = 0; j < 8; ++j)
#pragma unroll
            for (int r = 0; r < 8; ++r) {
                int m = wave * 16 + r + hi * 8;
                int n = j * 16 + lo;
                float v = acc[j][r] + biasL[L][n];
                zout[m * HD + n] = (half_t)fmaxf(v, 0.f);
            }
        {
            const int4* s4 = (const int4*)stageW[L];
            int4* d4 = (int4*)lds_w;
#pragma unroll
            for (int i = 0; i < 8; ++i) d4[tid + 256 * i] = s4[tid + 256 * i];
        }
        __syncthreads();
        half_t* t = zin; zin = zout; zout = t;
    }

    // ---- layer 4 (no relu) + atomic scatter to msg[dst] ----
#pragma unroll
    for (int j = 0; j < 8; ++j)
#pragma unroll
        for (int r = 0; r < 8; ++r) acc[j][r] = 0.f;
    for (int kk = 0; kk < HD; kk += 32) {
        v16h a = frag_a16(zin, HD, wave * 16, kk, lo, hi);
#pragma unroll
        for (int j = 0; j < 8; ++j) {
            v16h b = frag_b16(lds_w, HD, j * 16, kk, lo, hi);
            acc[j] = wmma_f16(a, b, acc[j]);
        }
    }
#pragma unroll
    for (int j = 0; j < 8; ++j)
#pragma unroll
        for (int r = 0; r < 8; ++r) {
            int m = wave * 16 + r + hi * 8;
            int n = j * 16 + lo;
            float v = acc[j][r] + b4[n];
            int d = dst[e0 + m];
            unsafeAtomicAdd(&msg[(size_t)d * HD + n], v);
        }
}

// ---------------- LSTM: gates = [cell_x|msg0..4|rnn_h] @ [Wih;Whh] + nonlinearity ----------------
// One block per 64 cell rows. K = 896, N = 512.

__global__ __launch_bounds__(256) void lstm_kernel(
    const half_t* __restrict__ cellx, const float* __restrict__ msgs,
    half_t* __restrict__ rnnh, float* __restrict__ rnnc,
    const half_t* __restrict__ Wgt /* [512][896] N-major */)
{
    __shared__ __align__(64) char smem[147456];
    half_t* lds_a = (half_t*)smem;             // 64 x 896 f16 = 114688 B
    half_t* lds_b = (half_t*)(smem + 114688);  // 512 x 32 f16 =  32768 B
    float*  lds_g = (float*)smem;              // 64 x 512 f32 = 131072 B (reuse)

    const int tid  = threadIdx.x;
    const int wave = tid >> 5;
    const int lane = tid & 31;
    const int lo   = lane & 15;
    const int hi   = lane >> 4;
    const int row0 = blockIdx.x * 64;
    const int mt   = wave >> 1;   // 0..3 : M tile
    const int nh   = wave & 1;    // 0..1 : N half

    // build A [64 x 896] f16
    for (int i = tid; i < 64 * 896; i += 256) {
        int r = i / 896, c = i - r * 896;
        int grow = row0 + r;
        float v;
        if (c < HD) {
            v = (float)cellx[(size_t)grow * HD + c];
        } else if (c < 768) {
            int t  = (c - HD) >> 7;
            int hh = c & 127;
            v = msgs[((size_t)t * NCELL + grow) * HD + hh];
        } else {
            v = (float)rnnh[(size_t)grow * HD + (c - 768)];
        }
        lds_a[i] = (half_t)v;
    }

    v8f acc[16];
#pragma unroll
    for (int j = 0; j < 16; ++j)
#pragma unroll
        for (int r = 0; r < 8; ++r) acc[j][r] = 0.f;

    for (int kk = 0; kk < 896; kk += 32) {
        __syncthreads();  // A build done (first iter) / previous B chunk consumed
        {
            int4* d4 = (int4*)lds_b;
            for (int i = tid; i < 2048; i += 256) {
                int n = i >> 2, qd = i & 3;
                d4[i] = *(const int4*)(Wgt + (size_t)n * 896 + kk + qd * 8);
            }
        }
        __syncthreads();
        v16h a = frag_a16(lds_a, 896, mt * 16, kk, lo, hi);
#pragma unroll
        for (int j = 0; j < 16; ++j) {
            v16h b = frag_b16(lds_b, 32, nh * 256 + j * 16, 0, lo, hi);
            acc[j] = wmma_f16(a, b, acc[j]);
        }
    }
    __syncthreads();

    // gates -> LDS f32 (reuses A/B space)
#pragma unroll
    for (int j = 0; j < 16; ++j)
#pragma unroll
        for (int r = 0; r < 8; ++r) {
            int m = mt * 16 + r + hi * 8;
            int n = nh * 256 + j * 16 + lo;
            lds_g[m * 512 + n] = acc[j][r];
        }
    __syncthreads();

    // LSTM nonlinearity (torch gate order i,f,g,o), in-place state update
    for (int i = tid; i < 64 * HD; i += 256) {
        int m = i >> 7, hh = i & 127;
        size_t g = (size_t)(row0 + m) * HD + hh;
        float iv = lds_g[m * 512 + hh];
        float fv = lds_g[m * 512 + 128 + hh];
        float gv = lds_g[m * 512 + 256 + hh];
        float ov = lds_g[m * 512 + 384 + hh];
        float cprev = rnnc[g];
        float cnew  = sigm(fv) * cprev + sigm(iv) * tanhf(gv);
        float hnew  = sigm(ov) * tanhf(cnew);
        rnnc[g] = cnew;
        rnnh[g] = (half_t)hnew;
    }
}

// ---------------- Logits: per-board [256 x 128] @ [128 x 17] via WMMA ----------------

__global__ __launch_bounds__(256) void logits_kernel(
    const half_t* __restrict__ h, const half_t* __restrict__ oemb,
    float* __restrict__ out)
{
    const int b    = blockIdx.x;
    const int tid  = threadIdx.x;
    const int wave = tid >> 5;
    const int lane = tid & 31;
    const int lo   = lane & 15;
    const int hi   = lane >> 4;
    const half_t* A  = h    + (size_t)b * 256 * HD;
    const half_t* Bm = oemb + (size_t)b * 17 * HD;

    v8f acc[2][2];
#pragma unroll
    for (int mi = 0; mi < 2; ++mi)
#pragma unroll
        for (int ni = 0; ni < 2; ++ni)
#pragma unroll
            for (int r = 0; r < 8; ++r) acc[mi][ni][r] = 0.f;

    for (int kk = 0; kk < HD; kk += 32) {
        v16h a0 = frag_a16(A, HD, (wave * 2 + 0) * 16, kk, lo, hi);
        v16h a1 = frag_a16(A, HD, (wave * 2 + 1) * 16, kk, lo, hi);
#pragma unroll
        for (int ni = 0; ni < 2; ++ni) {
            int c = ni * 16 + lo;
            if (c > 16) c = 16;  // clamp padded columns (masked at store)
            v16h bf = *(const v16h*)(Bm + (size_t)c * HD + kk + hi * 16);
            acc[0][ni] = wmma_f16(a0, bf, acc[0][ni]);
            acc[1][ni] = wmma_f16(a1, bf, acc[1][ni]);
        }
    }
#pragma unroll
    for (int mi = 0; mi < 2; ++mi)
#pragma unroll
        for (int ni = 0; ni < 2; ++ni)
#pragma unroll
            for (int r = 0; r < 8; ++r) {
                int m = (wave * 2 + mi) * 16 + r + hi * 8;
                int c = ni * 16 + lo;
                if (c < 17) out[((size_t)b * 256 + m) * 17 + c] = acc[mi][ni][r];
            }
}

// ---------------- prep kernels ----------------

__global__ void k_avg(const float* __restrict__ emb, float* __restrict__ avg) {
    int i = blockIdx.x * blockDim.x + threadIdx.x;
    if (i >= NB * HD) return;
    int b = i >> 7, hh = i & 127;
    float s = 0.f;
    for (int c = 1; c <= 16; ++c) s += emb[((size_t)b * 17 + c) * HD + hh];
    avg[i] = s * (1.f / 16.f);
}

__global__ void k_cellx(const float* __restrict__ emb, const float* __restrict__ avg,
                        const int* __restrict__ q, half_t* __restrict__ cellx) {
    int i = blockIdx.x * blockDim.x + threadIdx.x;
    if (i >= NCELL * HD) return;
    int cell = i >> 7, hh = i & 127;
    int b = cell >> 8;
    int qv = q[cell];
    float v = (qv == 0) ? avg[b * HD + hh] : emb[((size_t)b * 17 + qv) * HD + hh];
    cellx[i] = (half_t)v;
}

__global__ void k_f32_to_f16(const float* __restrict__ s, half_t* __restrict__ d, int n) {
    int i = blockIdx.x * blockDim.x + threadIdx.x;
    if (i < n) d[i] = (half_t)s[i];
}

__global__ void k_w1t(const float* __restrict__ W1, half_t* __restrict__ W1t) {
    int i = blockIdx.x * blockDim.x + threadIdx.x;
    if (i >= NT * 128 * 256) return;
    int t = i / (128 * 256);
    int rem = i - t * 128 * 256;
    int n = rem >> 8;
    int k = rem & 255;
    W1t[i] = (half_t)W1[((size_t)t * 256 + k) * 128 + n];
}

__global__ void k_w128t(const float* __restrict__ W, half_t* __restrict__ Wt) {
    int i = blockIdx.x * blockDim.x + threadIdx.x;
    if (i >= NT * 128 * 128) return;
    int t = i >> 14;
    int rem = i & 16383;
    int n = rem >> 7;
    int k = rem & 127;
    Wt[i] = (half_t)W[((size_t)t * 128 + k) * 128 + n];
}

__global__ void k_wgt(const float* __restrict__ Wih, const float* __restrict__ Whh,
                      half_t* __restrict__ Wgt) {
    int i = blockIdx.x * blockDim.x + threadIdx.x;
    if (i >= 512 * 896) return;
    int n = i / 896, k = i - n * 896;
    float v = (k < 768) ? Wih[(size_t)k * 512 + n] : Whh[(size_t)(k - 768) * 512 + n];
    Wgt[i] = (half_t)v;
}

// ---------------- host launch ----------------

extern "C" void kernel_launch(void* const* d_in, const int* in_sizes, int n_in,
                              void* d_out, int out_size, void* d_ws, size_t ws_size,
                              hipStream_t stream) {
    (void)in_sizes; (void)n_in; (void)out_size; (void)ws_size;
    const int*   edges  = (const int*)d_in[0];
    const int*   q      = (const int*)d_in[1];
    const float* in_emb = (const float*)d_in[2];
    const float* o_emb  = (const float*)d_in[3];
    const float* W1 = (const float*)d_in[4];
    const float* b1 = (const float*)d_in[5];
    const float* W2 = (const float*)d_in[6];
    const float* b2 = (const float*)d_in[7];
    const float* W3 = (const float*)d_in[8];
    const float* b3 = (const float*)d_in[9];
    const float* W4 = (const float*)d_in[10];
    const float* b4 = (const float*)d_in[11];
    const float* Wih = (const float*)d_in[12];
    const float* Whh = (const float*)d_in[13];
    float* out = (float*)d_out;

    char* ws = (char*)d_ws;
    size_t off = 0;
    auto alloc = [&](size_t bytes) -> char* {
        char* p = ws + off;
        off = (off + bytes + 255) & ~(size_t)255;
        return p;
    };
    half_t* cellx = (half_t*)alloc((size_t)NCELL * HD * 2);
    half_t* rnnh  = (half_t*)alloc((size_t)NCELL * HD * 2);
    half_t* hclu  = (half_t*)alloc((size_t)NCLU * HD * 2);
    half_t* oembh = (half_t*)alloc((size_t)NCLU * HD * 2);
    float*  rnnc  = (float*)alloc((size_t)NCELL * HD * 4);
    float*  msgs  = (float*)alloc((size_t)NT * NCELL * HD * 4);
    float*  avg   = (float*)alloc((size_t)NB * HD * 4);
    half_t* W1t   = (half_t*)alloc((size_t)NT * 128 * 256 * 2);
    half_t* W2t   = (half_t*)alloc((size_t)NT * 128 * 128 * 2);
    half_t* W3t   = (half_t*)alloc((size_t)NT * 128 * 128 * 2);
    half_t* W4t   = (half_t*)alloc((size_t)NT * 128 * 128 * 2);
    half_t* Wgt   = (half_t*)alloc((size_t)512 * 896 * 2);

    hipMemsetAsync(rnnh, 0, (size_t)NCELL * HD * 2, stream);
    hipMemsetAsync(rnnc, 0, (size_t)NCELL * HD * 4, stream);

    k_avg<<<(NB * HD + 255) / 256, 256, 0, stream>>>(in_emb, avg);
    k_cellx<<<(NCELL * HD + 255) / 256, 256, 0, stream>>>(in_emb, avg, q, cellx);
    k_f32_to_f16<<<(NCLU * HD + 255) / 256, 256, 0, stream>>>(in_emb, hclu, NCLU * HD);
    k_f32_to_f16<<<(NCLU * HD + 255) / 256, 256, 0, stream>>>(o_emb, oembh, NCLU * HD);
    k_w1t<<<(NT * 128 * 256 + 255) / 256, 256, 0, stream>>>(W1, W1t);
    k_w128t<<<(NT * 128 * 128 + 255) / 256, 256, 0, stream>>>(W2, W2t);
    k_w128t<<<(NT * 128 * 128 + 255) / 256, 256, 0, stream>>>(W3, W3t);
    k_w128t<<<(NT * 128 * 128 + 255) / 256, 256, 0, stream>>>(W4, W4t);
    k_wgt<<<(512 * 896 + 255) / 256, 256, 0, stream>>>(Wih, Whh, Wgt);

    for (int s = 0; s < NS; ++s) {
        hipMemsetAsync(msgs, 0, (size_t)NT * NCELL * HD * 4, stream);
        const half_t* hcur = (s == 0) ? cellx : rnnh;
        for (int t = 0; t < NT; ++t) {
            const int* srcp = edges + ((size_t)t * 2 + 0) * ET;
            const int* dstp = edges + ((size_t)t * 2 + 1) * ET;
            const half_t* hs = (t == 0) ? hclu : hcur;
            edge_mlp_kernel<<<ET / 128, 256, 0, stream>>>(
                srcp, dstp, hs, hcur,
                W1t + (size_t)t * 128 * 256, W2t + (size_t)t * 128 * 128,
                W3t + (size_t)t * 128 * 128, W4t + (size_t)t * 128 * 128,
                b1 + t * HD, b2 + t * HD, b3 + t * HD, b4 + t * HD,
                msgs + (size_t)t * NCELL * HD);
        }
        lstm_kernel<<<NCELL / 64, 256, 0, stream>>>(cellx, msgs, rnnh, rnnc, Wgt);
        logits_kernel<<<NB, 256, 0, stream>>>(rnnh, oembh, out + (size_t)s * NCELL * 17);
    }
}
